// TransformerModel_20856361189764
// MI455X (gfx1250) — compile-verified, matrix operand order checked by reading, not run on hardware
//
#include <hip/hip_runtime.h>
#include <hip/hip_bf16.h>
#include <climits>

// ---------------- Static problem dims (match reference) ----------------
#define NNODES 50000
#define NEDGES 800000
#define DIN0   64
#define HC     128          // H*C
#define NH     2
#define CH     64
#define EDIMS  14
#define NG     256
#define FCH    64
#define QKVSW  512          // Q|K|V|S fused output width

typedef __attribute__((ext_vector_type(16))) __bf16 v16bf;
typedef __attribute__((ext_vector_type(8)))  float  v8f;

// ---------------- small device helpers ----------------
static __device__ __forceinline__ unsigned short f32_to_bf16(float f) {
  unsigned int u = __float_as_uint(f);
  unsigned int r = u + 0x7fffu + ((u >> 16) & 1u);   // round-to-nearest-even
  return (unsigned short)(r >> 16);
}
// order-preserving float<->int encoding (for atomicMax on signed values)
static __device__ __forceinline__ int enc_ord(float f) {
  int i = __float_as_int(f);
  return (i >= 0) ? i : (i ^ 0x7fffffff);
}
static __device__ __forceinline__ float dec_ord(int i) {
  return __int_as_float((i >= 0) ? i : (i ^ 0x7fffffff));
}

// ---------------- fp32 -> bf16 conversion ----------------
__global__ void cvt_bf16_kernel(const float* __restrict__ src,
                                unsigned short* __restrict__ dst, int n) {
  int gid = blockIdx.x * 256 + threadIdx.x;
  if (gid < n) dst[gid] = f32_to_bf16(src[gid]);
}

// ---------------- pack 4 weight mats [din,128] -> Wt[512,din] bf16 (transposed), bias[512] ----------------
__global__ void pack_wb_kernel(const float* __restrict__ Wq, const float* __restrict__ Wk,
                               const float* __restrict__ Wv, const float* __restrict__ Ws,
                               const float* __restrict__ bq, const float* __restrict__ bk,
                               const float* __restrict__ bv, const float* __restrict__ bs,
                               unsigned short* __restrict__ Wt, float* __restrict__ bp, int din) {
  int gid = blockIdx.x * 256 + threadIdx.x;
  int total = QKVSW * din;
  if (gid < total) {
    int n  = gid / din;
    int k  = gid - n * din;
    int mi = n >> 7;          // 0:Q 1:K 2:V 3:S
    int c  = n & 127;
    const float* W = (mi == 0) ? Wq : (mi == 1) ? Wk : (mi == 2) ? Wv : Ws;
    Wt[(size_t)n * din + k] = f32_to_bf16(W[(size_t)k * HC + c]);
  }
  if (gid < QKVSW) {
    int mi = gid >> 7, c = gid & 127;
    const float* B = (mi == 0) ? bq : (mi == 1) ? bk : (mi == 2) ? bv : bs;
    bp[gid] = B[c];
  }
}

// ---------------- fused Q|K|V|S GEMM: [N,din]bf16 x [din,512]bf16 -> [N,512]f32 via WMMA ----------------
// 128 threads = 4 waves per block. blockIdx.x -> 16-row M tile; wave w handles a group of
// FOUR 16-col N tiles (blockIdx.y*4 + w selects the group), so the A fragment is loaded
// once per k-step and reused by 4 independent back-to-back v_wmma (no D->A/B RAW hazard).
__global__ void gemm_qkvs_wmma_kernel(const unsigned short* __restrict__ Xbf,
                                      const unsigned short* __restrict__ Wt,
                                      const float* __restrict__ bp,
                                      float* __restrict__ out, int din) {
  const int wave  = threadIdx.x >> 5;
  const int lane  = threadIdx.x & 31;
  const int m     = lane & 15;     // A row / B col within tile
  const int half  = lane >> 4;     // K half selector
  const int mbase = blockIdx.x * 16;
  const int nbase = (blockIdx.y * 4 + wave) * 64;   // 4 tiles of 16 cols
  const size_t arow = (size_t)(mbase + m) * din;
  const size_t brow = (size_t)(nbase + m) * din;    // Wt is transposed: row = out col
  const size_t bstr = (size_t)16 * din;             // next N tile

  v8f acc0 = {}, acc1 = {}, acc2 = {}, acc3 = {};
  const int ksteps = din >> 5;     // K in chunks of 32
  for (int kt = 0; kt < ksteps; ++kt) {
    const int k0 = kt * 32;
    union { v16bf v; uint4 q[2]; } a, b0, b1, b2, b3;
    // A 16x32 bf16 fragment (ISA 7.12.2): lane(m,half):
    //   a[0..7]=A[m][k0+half*8+i], a[8..15]=A[m][k0+16+half*8+i]
    a.q[0] = *reinterpret_cast<const uint4*>(Xbf + arow + k0 + half * 8);
    a.q[1] = *reinterpret_cast<const uint4*>(Xbf + arow + k0 + 16 + half * 8);
    // 4 B fragments (column-striped, symmetric to A over columns)
    const unsigned short* wp = Wt + brow + k0 + half * 8;
    b0.q[0] = *reinterpret_cast<const uint4*>(wp);
    b0.q[1] = *reinterpret_cast<const uint4*>(wp + 16);
    b1.q[0] = *reinterpret_cast<const uint4*>(wp + bstr);
    b1.q[1] = *reinterpret_cast<const uint4*>(wp + bstr + 16);
    b2.q[0] = *reinterpret_cast<const uint4*>(wp + 2 * bstr);
    b2.q[1] = *reinterpret_cast<const uint4*>(wp + 2 * bstr + 16);
    b3.q[0] = *reinterpret_cast<const uint4*>(wp + 3 * bstr);
    b3.q[1] = *reinterpret_cast<const uint4*>(wp + 3 * bstr + 16);
    acc0 = __builtin_amdgcn_wmma_f32_16x16x32_bf16(false, a.v, false, b0.v, (short)0, acc0, false, false);
    acc1 = __builtin_amdgcn_wmma_f32_16x16x32_bf16(false, a.v, false, b1.v, (short)0, acc1, false, false);
    acc2 = __builtin_amdgcn_wmma_f32_16x16x32_bf16(false, a.v, false, b2.v, (short)0, acc2, false, false);
    acc3 = __builtin_amdgcn_wmma_f32_16x16x32_bf16(false, a.v, false, b3.v, (short)0, acc3, false, false);
  }
  // D layout: VGPR r -> M = r + 8*half, N = lane&15
  const v8f accs[4] = {acc0, acc1, acc2, acc3};
#pragma unroll
  for (int t = 0; t < 4; ++t) {
    const int col  = nbase + t * 16 + m;
    const float bias = bp[col];
#pragma unroll
    for (int r = 0; r < 8; ++r) {
      out[(size_t)(mbase + half * 8 + r) * QKVSW + col] = accs[t][r] + bias;
    }
  }
}

// ---------------- per-layer accumulator init ----------------
__global__ void init_layer_kernel(float* __restrict__ num, int* __restrict__ menc,
                                  float* __restrict__ den) {
  int gid = blockIdx.x * 256 + threadIdx.x;
  if (gid < NNODES * HC) num[gid] = 0.f;
  if (gid < NNODES * NH) { menc[gid] = INT_MIN; den[gid] = 0.f; }
}

// ---------------- edge pass A: scores + segment max ----------------
__global__ void edge_scores_kernel(const float* __restrict__ EA, const int* __restrict__ EI,
                                   const float* __restrict__ We, const float* __restrict__ QKVS,
                                   float* __restrict__ scores, int* __restrict__ menc) {
  __shared__ float sWe[EDIMS * HC];                // 7 KB of the 320 KB WGP LDS
  for (int i = threadIdx.x; i < EDIMS * HC; i += 256) sWe[i] = We[i];
  __syncthreads();
  int gid = blockIdx.x * 256 + threadIdx.x;
  if (gid >= NEDGES * NH) return;
  int e = gid >> 1, h = gid & 1;
  int src = EI[e], dst = EI[NEDGES + e];
  const float* Kr = QKVS + (size_t)src * QKVSW + 128 + h * CH;
  const float* Qr = QKVS + (size_t)dst * QKVSW + h * CH;
  __builtin_prefetch(Kr, 0, 0);                    // global_prefetch: hide L2 gather latency
  __builtin_prefetch(Qr, 0, 0);                    // behind edge_attr loads + LDS reads
  float ea[EDIMS];
#pragma unroll
  for (int d = 0; d < EDIMS; ++d) ea[d] = EA[(size_t)e * EDIMS + d];
  const float4* K4 = reinterpret_cast<const float4*>(Kr);
  const float4* Q4 = reinterpret_cast<const float4*>(Qr);
  const float* w = sWe + h * CH;
  float s = 0.f;
#pragma unroll 4
  for (int cc = 0; cc < CH / 4; ++cc) {
    float4 kv = K4[cc], qv = Q4[cc];
    const float* kp = reinterpret_cast<const float*>(&kv);
    const float* qp = reinterpret_cast<const float*>(&qv);
#pragma unroll
    for (int j = 0; j < 4; ++j) {
      int c = cc * 4 + j;
      float ek = 0.f;
#pragma unroll
      for (int d = 0; d < EDIMS; ++d) ek = fmaf(ea[d], w[d * HC + c], ek);
      s = fmaf(qp[j], kp[j] + ek, s);
    }
  }
  s *= 0.125f;                                    // 1/sqrt(C), C=64
  scores[gid] = s;
  atomicMax(menc + dst * NH + h, enc_ord(s));
}

// ---------------- edge pass B: exp, denominator, numerator (= sum ex * value) ----------------
__global__ void edge_accum_kernel(const float* __restrict__ EA, const int* __restrict__ EI,
                                  const float* __restrict__ We, const float* __restrict__ QKVS,
                                  const float* __restrict__ scores, const int* __restrict__ menc,
                                  float* __restrict__ den, float* __restrict__ num) {
  __shared__ float sWe[EDIMS * HC];
  for (int i = threadIdx.x; i < EDIMS * HC; i += 256) sWe[i] = We[i];
  __syncthreads();
  int gid = blockIdx.x * 256 + threadIdx.x;
  if (gid >= NEDGES * NH) return;
  int e = gid >> 1, h = gid & 1;
  int src = EI[e], dst = EI[NEDGES + e];
  const float* Vr = QKVS + (size_t)src * QKVSW + 256 + h * CH;
  __builtin_prefetch(Vr, 0, 0);
  float mx = dec_ord(menc[dst * NH + h]);
  float ex = __expf(scores[gid] - mx);
  atomicAdd(den + dst * NH + h, ex);
  float ea[EDIMS];
#pragma unroll
  for (int d = 0; d < EDIMS; ++d) ea[d] = EA[(size_t)e * EDIMS + d];
  const float4* V4 = reinterpret_cast<const float4*>(Vr);
  const float* w = sWe + h * CH;
  float* np = num + (size_t)dst * HC + h * CH;
#pragma unroll 4
  for (int cc = 0; cc < CH / 4; ++cc) {
    float4 vv = V4[cc];
    const float* vp = reinterpret_cast<const float*>(&vv);
#pragma unroll
    for (int j = 0; j < 4; ++j) {
      int c = cc * 4 + j;
      float ek = 0.f;
#pragma unroll
      for (int d = 0; d < EDIMS; ++d) ek = fmaf(ea[d], w[d * HC + c], ek);
      atomicAdd(np + c, ex * (vp[j] + ek));
    }
  }
}

// ---------------- node finalize: out = relu(num/den + skip) ----------------
__global__ void node_finalize_kernel(const float* __restrict__ num, const float* __restrict__ den,
                                     const float* __restrict__ QKVS, float* __restrict__ hbuf) {
  int gid = blockIdx.x * 256 + threadIdx.x;
  if (gid >= NNODES * HC) return;
  int node = gid >> 7, ch = gid & 127, h = ch >> 6;
  float d = den[node * NH + h];
  float v = num[gid] / (d + 1e-16f) + QKVS[(size_t)node * QKVSW + 384 + ch];
  hbuf[gid] = fmaxf(v, 0.f);
}

// ---------------- global max pool over batch groups ----------------
__global__ void pool_init_kernel(int* __restrict__ genc) {
  int gid = blockIdx.x * 256 + threadIdx.x;
  if (gid < NG * HC) genc[gid] = INT_MIN;
}
__global__ void pool_max_kernel(const float* __restrict__ hbuf, const int* __restrict__ batch,
                                int* __restrict__ genc) {
  int gid = blockIdx.x * 256 + threadIdx.x;
  if (gid >= NNODES * HC) return;
  int node = gid >> 7, ch = gid & 127;
  atomicMax(genc + batch[node] * HC + ch, enc_ord(hbuf[gid]));
}

// ---------------- FC head: Linear(128->64) + ReLU + BN(eval) + Linear(64->1) ----------------
__global__ void fc_head_kernel(const int* __restrict__ genc,
                               const float* __restrict__ W1, const float* __restrict__ b1,
                               const float* __restrict__ bng, const float* __restrict__ bnb,
                               const float* __restrict__ bnm, const float* __restrict__ bnv,
                               const float* __restrict__ W2, const float* __restrict__ b2,
                               float* __restrict__ out) {
  __shared__ float grow[HC];
  __shared__ float red[FCH];
  int g = blockIdx.x, tid = threadIdx.x;
  for (int i = tid; i < HC; i += FCH) grow[i] = dec_ord(genc[g * HC + i]);
  __syncthreads();
  float z = b1[tid];
#pragma unroll 8
  for (int k = 0; k < HC; ++k) z = fmaf(grow[k], W1[k * FCH + tid], z);
  z = fmaxf(z, 0.f);
  z = bng[tid] * (z - bnm[tid]) * rsqrtf(bnv[tid] + 1e-5f) + bnb[tid];
  red[tid] = z * W2[tid];
  __syncthreads();
  for (int off = FCH / 2; off > 0; off >>= 1) {
    if (tid < off) red[tid] += red[tid + off];
    __syncthreads();
  }
  if (tid == 0) out[g] = red[0] + b2[0];
}

// =======================================================================
extern "C" void kernel_launch(void* const* d_in, const int* in_sizes, int n_in,
                              void* d_out, int out_size, void* d_ws, size_t ws_size,
                              hipStream_t stream) {
  (void)in_sizes; (void)n_in; (void)out_size; (void)ws_size;

  const float* x     = (const float*)d_in[0];
  const float* EA    = (const float*)d_in[1];
  const int*   EI    = (const int*)d_in[2];
  const int*   batch = (const int*)d_in[3];
  // layer param blocks: l0 -> d_in[4..12], l1 -> d_in[13..21]
  const float* Wq[2] = {(const float*)d_in[4],  (const float*)d_in[13]};
  const float* bq[2] = {(const float*)d_in[5],  (const float*)d_in[14]};
  const float* Wk[2] = {(const float*)d_in[6],  (const float*)d_in[15]};
  const float* bk[2] = {(const float*)d_in[7],  (const float*)d_in[16]};
  const float* Wv[2] = {(const float*)d_in[8],  (const float*)d_in[17]};
  const float* bv[2] = {(const float*)d_in[9],  (const float*)d_in[18]};
  const float* Ws[2] = {(const float*)d_in[10], (const float*)d_in[19]};
  const float* bs[2] = {(const float*)d_in[11], (const float*)d_in[20]};
  const float* We[2] = {(const float*)d_in[12], (const float*)d_in[21]};
  const float* fcW1  = (const float*)d_in[22];
  const float* fcb1  = (const float*)d_in[23];
  const float* bn_g  = (const float*)d_in[24];
  const float* bn_b  = (const float*)d_in[25];
  const float* bn_m  = (const float*)d_in[26];
  const float* bn_v  = (const float*)d_in[27];
  const float* fcW2  = (const float*)d_in[28];
  const float* fcb2  = (const float*)d_in[29];

  // ---- workspace carve-up (256B aligned) ----
  char* ws = (char*)d_ws;
  size_t off = 0;
  auto carve = [&](size_t bytes) { size_t o = off; off += (bytes + 255) & ~(size_t)255; return o; };
  unsigned short* xbf   = (unsigned short*)(ws + carve((size_t)NNODES * HC * 2));      // 12.8 MB
  unsigned short* wt    = (unsigned short*)(ws + carve((size_t)QKVSW  * HC * 2));      // 131 KB
  float*          bp    = (float*)(ws + carve((size_t)QKVSW * 4));                     // 2 KB
  float*          qkvs  = (float*)(ws + carve((size_t)NNODES * QKVSW * 4));            // 102.4 MB (L2-resident)
  float*          scbuf = (float*)(ws + carve((size_t)NEDGES * NH * 4));               // 6.4 MB
  int*            menc  = (int*)  (ws + carve((size_t)NNODES * NH * 4));               // 0.4 MB
  float*          den   = (float*)(ws + carve((size_t)NNODES * NH * 4));               // 0.4 MB
  float*          num   = (float*)(ws + carve((size_t)NNODES * HC * 4));               // 25.6 MB
  float*          hbuf  = (float*)(ws + carve((size_t)NNODES * HC * 4));               // 25.6 MB
  int*            genc  = (int*)  (ws + carve((size_t)NG * HC * 4));                   // 128 KB

  const int blkE  = (NEDGES * NH + 255) / 256;   // 6250
  const int blkNC = (NNODES * HC + 255) / 256;   // 25000

  for (int l = 0; l < 2; ++l) {
    const int din = (l == 0) ? DIN0 : HC;
    const float* xin = (l == 0) ? x : hbuf;

    pack_wb_kernel<<<(QKVSW * din + 255) / 256, 256, 0, stream>>>(
        Wq[l], Wk[l], Wv[l], Ws[l], bq[l], bk[l], bv[l], bs[l], wt, bp, din);
    cvt_bf16_kernel<<<(NNODES * din + 255) / 256, 256, 0, stream>>>(xin, xbf, NNODES * din);
    gemm_qkvs_wmma_kernel<<<dim3(NNODES / 16, 2), 128, 0, stream>>>(xbf, wt, bp, qkvs, din);
    init_layer_kernel<<<blkNC, 256, 0, stream>>>(num, menc, den);
    edge_scores_kernel<<<blkE, 256, 0, stream>>>(EA, EI, We[l], qkvs, scbuf, menc);
    edge_accum_kernel<<<blkE, 256, 0, stream>>>(EA, EI, We[l], qkvs, scbuf, menc, den, num);
    node_finalize_kernel<<<blkNC, 256, 0, stream>>>(num, den, qkvs, hbuf);
  }

  pool_init_kernel<<<(NG * HC + 255) / 256, 256, 0, stream>>>(genc);
  pool_max_kernel<<<blkNC, 256, 0, stream>>>(hbuf, batch, genc);
  fc_head_kernel<<<NG, FCH, 0, stream>>>(genc, fcW1, fcb1, bn_g, bn_b, bn_m, bn_v,
                                         fcW2, fcb2, (float*)d_out);
}
